// RSNN_22643067584688
// MI455X (gfx1250) — matrix operand order repaired and checked by reading, not compiled, not run on hardware
//
#include <hip/hip_runtime.h>

// ---------------------------------------------------------------------------
// RSNN forward on gfx1250 (MI455X): bf16 WMMA GEMMs + fused LIF dynamics.
//   B=128, T=500, I=700 (pad 704), H=1024, O=20 (pad 32)
// d_out layout (float32, concatenated): out_rec(B,501,20) | mem_rec(B,501,1024)
//                                       | spk_rec(B,501,1024)
// Hot GEMMs: double-buffered LDS staged with GLOBAL_LOAD_ASYNC_TO_LDS_B128
// (ASYNCcnt path). K-loop manually unrolled by 2 (NK even) so both ping-pong
// bodies accumulate into the SAME registers -> no cross-body accumulator
// copies, and staging addresses advance by pointer increments.
// ---------------------------------------------------------------------------

typedef __attribute__((ext_vector_type(16))) __bf16 v16bf;
typedef __attribute__((ext_vector_type(8)))  __bf16 v8bf;
typedef __attribute__((ext_vector_type(8)))  float  v8f;
typedef __attribute__((ext_vector_type(4)))  float  v4f;

#define B_    128
#define T_    500
#define TP1   501
#define I_    700
#define IPAD  704
#define H_    1024
#define O_    20
#define OPAD  32
#define LS    40     // LDS row stride (bf16 elems): 32 data + 8 pad (bank skew)

union Fr16 { v16bf v; v8bf h[2]; };

static __device__ __forceinline__ v8f wmma_bf16(v16bf a, v16bf b, v8f c) {
  // D = A(16x32 bf16) x B(32x16 bf16) + C(16x16 f32)
  return __builtin_amdgcn_wmma_f32_16x16x32_bf16(
      /*neg_a=*/false, a, /*neg_b=*/false, b,
      /*c_mod=*/(short)0, c, /*reuse_a=*/false, /*reuse_b=*/false);
}

// Low 32 bits of a generic LDS pointer == wave-relative LDS byte offset
// (ISA 10.2 aperture mapping), which is what the async VDST operand wants.
static __device__ __forceinline__ unsigned lds_addr32(const void* p) {
  return (unsigned)(uintptr_t)p;
}

// Per-lane async copy of 16 bytes global -> LDS (ASYNCcnt-tracked, no VGPRs).
static __device__ __forceinline__ void async_ld16(unsigned lds_off,
                                                  const void* gaddr) {
  asm volatile("global_load_async_to_lds_b128 %0, %1, off"
               :
               : "v"(lds_off), "v"((unsigned long long)(uintptr_t)gaddr)
               : "memory");
}

static __device__ __forceinline__ void s_wait_async_le4() {
  asm volatile("s_wait_asynccnt 4" ::: "memory");
}
static __device__ __forceinline__ void s_wait_async_0() {
  asm volatile("s_wait_asynccnt 0" ::: "memory");
}

// 16x64 strip of C per wave: A-frag (reused 4x) * 4 B-frags, in-place accum.
static __device__ __forceinline__ void compute_tile4(const __bf16* Ab,
                                                     const __bf16* Bb,
                                                     int wv, int l15, int hs,
                                                     v8f acc[4]) {
  Fr16 a;  // ISA 16-bit A layout: lane<16 -> K 0..7 & 16..23; lane>=16 -> +8
  a.h[0] = *(const v8bf*)(Ab + (wv * 16 + l15) * LS + hs * 8);
  a.h[1] = *(const v8bf*)(Ab + (wv * 16 + l15) * LS + 16 + hs * 8);
#pragma unroll
  for (int j = 0; j < 4; j++) {
    Fr16 b;  // B layout matches A: lane holds column n=l15, K hs*8..+7 & 16+hs*8..+7
    b.h[0] = *(const v8bf*)(Bb + (j * 16 + l15) * LS + hs * 8);
    b.h[1] = *(const v8bf*)(Bb + (j * 16 + l15) * LS + 16 + hs * 8);
    acc[j] = wmma_bf16(a.v, b.v, acc[j]);
  }
}

// --------------------------- precision converts -----------------------------

__global__ void cvt_inputs_kernel(const float* __restrict__ in,
                                  __bf16* __restrict__ xbf) {
  long long idx = (long long)blockIdx.x * blockDim.x + threadIdx.x;
  if (idx >= (long long)(B_ * T_) * IPAD) return;
  int r = (int)(idx / IPAD);
  int i = (int)(idx - (long long)r * IPAD);
  float v = (i < I_) ? in[(size_t)r * I_ + i] : 0.0f;
  xbf[idx] = (__bf16)v;
}

__global__ void cvt_w1_kernel(const float* __restrict__ w1,
                              __bf16* __restrict__ w1t) {
  int idx = blockIdx.x * blockDim.x + threadIdx.x;
  if (idx >= H_ * IPAD) return;
  int n = idx / IPAD, k = idx - n * IPAD;         // w1t[n][k] = w1[k][n]
  w1t[idx] = (__bf16)((k < I_) ? w1[(size_t)k * H_ + n] : 0.0f);
}

__global__ void cvt_v1_kernel(const float* __restrict__ v1,
                              __bf16* __restrict__ v1t) {
  int idx = blockIdx.x * blockDim.x + threadIdx.x;
  if (idx >= H_ * H_) return;
  int n = idx >> 10, k = idx & (H_ - 1);          // v1t[n][k] = v1m[k][n]
  v1t[idx] = (__bf16)((k == n) ? 0.0f : v1[(size_t)k * H_ + n]);
}

__global__ void cvt_w2_kernel(const float* __restrict__ w2,
                              __bf16* __restrict__ w2t) {
  int idx = blockIdx.x * blockDim.x + threadIdx.x;
  if (idx >= OPAD * H_) return;
  int n = idx >> 10, k = idx & (H_ - 1);          // w2t[n][k] = w2[k][n]
  w2t[idx] = (__bf16)((n < O_) ? w2[(size_t)k * O_ + n] : 0.0f);
}

// ------------------------------ init state ----------------------------------

__global__ void init_kernel(float* __restrict__ out_rec,
                            float* __restrict__ mem_rec,
                            float* __restrict__ spk_rec,
                            float* __restrict__ syn,
                            __bf16* __restrict__ h1bf0) {
  int idx = blockIdx.x * blockDim.x + threadIdx.x;
  if (idx < B_ * H_) {
    int bb = idx / H_, h = idx - bb * H_;
    mem_rec[(size_t)bb * (TP1 * H_) + h] = 0.0f;   // mem_rec[:,0,:]
    spk_rec[(size_t)bb * (TP1 * H_) + h] = 0.0f;   // spk_rec[:,0,:]
    syn[idx] = 0.0f;
    h1bf0[idx] = (__bf16)0.0f;
  }
  if (idx < B_ * O_) {
    int bb = idx / O_, o = idx - bb * O_;
    out_rec[(size_t)bb * (TP1 * O_) + o] = 0.0f;   // out_rec[:,0,:]
  }
}

// ---------------- feedforward GEMM: xft(T,B,H) = inputs @ w1 ----------------
// grid = (BT/64) * (H/64), block = 128 (4 waves), wave tile 16x64, K-step 32.

__global__ __launch_bounds__(128) void ff_gemm_kernel(
    const __bf16* __restrict__ xbf,   // [B*T][IPAD]   row r = b*T + t
    const __bf16* __restrict__ w1t,   // [H][IPAD]     (transposed, padded)
    float* __restrict__ xft)          // [T][B][H]
{
  __shared__ __align__(16) __bf16 Al[2][64 * LS];
  __shared__ __align__(16) __bf16 Bl[2][64 * LS];
  const int wg = blockIdx.x;
  const int Rbase = (wg >> 4) * 64;
  const int Nbase = (wg & 15) * 64;
  const int tid = threadIdx.x;
  const int lane = tid & 31, wv = tid >> 5;
  const int hs = lane >> 4, l15 = lane & 15;
  const int srow = tid >> 1, shh = tid & 1;

  // Per-thread staging addresses (16 contiguous bf16 = 2 x b128 per operand).
  const char* gA = (const char*)(xbf + (size_t)(Rbase + srow) * IPAD + shh * 16);
  const char* gB = (const char*)(w1t + (size_t)(Nbase + srow) * IPAD + shh * 16);
  const unsigned sA0 = lds_addr32(&Al[0][srow * LS + shh * 16]);
  const unsigned sA1 = lds_addr32(&Al[1][srow * LS + shh * 16]);
  const unsigned sB0 = lds_addr32(&Bl[0][srow * LS + shh * 16]);
  const unsigned sB1 = lds_addr32(&Bl[1][srow * LS + shh * 16]);

  v8f acc[4] = {};

  // Prologue: stage K-step 0 into buffer 0; gA/gB then point at next K-step.
  async_ld16(sA0, gA);  async_ld16(sA0 + 16, gA + 16);
  async_ld16(sB0, gB);  async_ld16(sB0 + 16, gB + 16);
  gA += 64; gB += 64;

  const int NK = IPAD / 32;   // 22 (even)
  for (int it = 0; it < NK; it += 2) {
    // ---- K-step it (buffer 0); prefetch it+1 into buffer 1 ----
    __syncthreads();                       // everyone done reading buf1
    async_ld16(sA1, gA);  async_ld16(sA1 + 16, gA + 16);
    async_ld16(sB1, gB);  async_ld16(sB1 + 16, gB + 16);
    gA += 64; gB += 64;
    s_wait_async_le4();                    // in-order => buf0 copies landed
    __syncthreads();                       // buf0 visible to all waves
    compute_tile4(Al[0], Bl[0], wv, l15, hs, acc);

    // ---- K-step it+1 (buffer 1); prefetch it+2 into buffer 0 ----
    __syncthreads();                       // everyone done reading buf0
    if (it + 2 < NK) {
      async_ld16(sA0, gA);  async_ld16(sA0 + 16, gA + 16);
      async_ld16(sB0, gB);  async_ld16(sB0 + 16, gB + 16);
      gA += 64; gB += 64;
      s_wait_async_le4();
    } else {
      s_wait_async_0();
    }
    __syncthreads();                       // buf1 visible to all waves
    compute_tile4(Al[1], Bl[1], wv, l15, hs, acc);
  }

#pragma unroll
  for (int j = 0; j < 4; j++) {
    int h = Nbase + j * 16 + l15;
#pragma unroll
    for (int v = 0; v < 8; v++) {
      int r = Rbase + wv * 16 + v + 8 * hs;     // r = b*T + t
      int bb = r / T_;
      int tt = r - bb * T_;
      xft[(size_t)tt * (B_ * H_) + (size_t)bb * H_ + h] = acc[j][v];
    }
  }
}

// ------------- recurrent step t: h1 = x_t + h1 @ v1m, LIF update ------------
// grid = (B/64)*(H/64) = 32, block = 128. C initialized with xft slab (the +x).

__global__ __launch_bounds__(128) void rec_step_kernel(
    int t,
    const float*  __restrict__ xft,      // [T][B][H]
    const __bf16* __restrict__ v1t,      // [H][H] transposed, diag zeroed
    const __bf16* __restrict__ h1in,     // [B][H] bf16 state (prev step)
    __bf16*       __restrict__ h1out,    // [B][H] bf16 state (this step)
    float*        __restrict__ syn,      // [B][H]
    const float*  __restrict__ alpha_h,
    const float*  __restrict__ beta_h,
    float*        __restrict__ mem_rec,  // [B][501][H]
    float*        __restrict__ spk_rec)  // [B][501][H]
{
  __shared__ __align__(16) __bf16 Al[2][64 * LS];
  __shared__ __align__(16) __bf16 Bl[2][64 * LS];
  const int wg = blockIdx.x;
  const int Mbase = (wg >> 4) * 64;   // batch tile
  const int Nbase = (wg & 15) * 64;   // h tile
  const int tid = threadIdx.x;
  const int lane = tid & 31, wv = tid >> 5;
  const int hs = lane >> 4, l15 = lane & 15;
  const int srow = tid >> 1, shh = tid & 1;

  const char* gA = (const char*)(h1in + (size_t)(Mbase + srow) * H_ + shh * 16);
  const char* gB = (const char*)(v1t  + (size_t)(Nbase + srow) * H_ + shh * 16);
  const unsigned sA0 = lds_addr32(&Al[0][srow * LS + shh * 16]);
  const unsigned sA1 = lds_addr32(&Al[1][srow * LS + shh * 16]);
  const unsigned sB0 = lds_addr32(&Bl[0][srow * LS + shh * 16]);
  const unsigned sB1 = lds_addr32(&Bl[1][srow * LS + shh * 16]);

  // Prologue stage (async copies fly while we init C from the drive slab).
  async_ld16(sA0, gA);  async_ld16(sA0 + 16, gA + 16);
  async_ld16(sB0, gB);  async_ld16(sB0 + 16, gB + 16);
  gA += 64; gB += 64;

  const float* xslab = xft + (size_t)t * (B_ * H_);
  v8f acc[4];
#pragma unroll
  for (int j = 0; j < 4; j++) {
    int h = Nbase + j * 16 + l15;
#pragma unroll
    for (int v = 0; v < 8; v++) {
      int bb = Mbase + wv * 16 + v + 8 * hs;
      acc[j][v] = xslab[(size_t)bb * H_ + h];
    }
  }

  const int NK = H_ / 32;   // 32 (even)
  for (int it = 0; it < NK; it += 2) {
    __syncthreads();
    async_ld16(sA1, gA);  async_ld16(sA1 + 16, gA + 16);
    async_ld16(sB1, gB);  async_ld16(sB1 + 16, gB + 16);
    gA += 64; gB += 64;
    s_wait_async_le4();
    __syncthreads();
    compute_tile4(Al[0], Bl[0], wv, l15, hs, acc);

    __syncthreads();
    if (it + 2 < NK) {
      async_ld16(sA0, gA);  async_ld16(sA0 + 16, gA + 16);
      async_ld16(sB0, gB);  async_ld16(sB0 + 16, gB + 16);
      gA += 64; gB += 64;
      s_wait_async_le4();
    } else {
      s_wait_async_0();
    }
    __syncthreads();
    compute_tile4(Al[1], Bl[1], wv, l15, hs, acc);
  }

  // Fused LIF epilogue straight from accumulator VGPRs.
#pragma unroll
  for (int j = 0; j < 4; j++) {
    int h = Nbase + j * 16 + l15;
    float al = alpha_h[h];
    float be = beta_h[h];
#pragma unroll
    for (int v = 0; v < 8; v++) {
      int bb = Mbase + wv * 16 + v + 8 * hs;
      float h1v = acc[j][v];                       // x_t + h1_prev @ v1m
      size_t sidx = (size_t)bb * H_ + h;
      size_t mrow = (size_t)bb * (TP1 * H_) + (size_t)t * H_ + h;
      float so = syn[sidx];
      float mo = mem_rec[mrow];
      float mthr = mo - 1.0f;
      float sp = (mthr > 0.0f) ? 1.0f : 0.0f;     // spike == reset
      mem_rec[mrow + H_] = be * mo + so - sp;     // mem_rec[:, t+1]
      spk_rec[mrow + H_] = sp;                    // spk_rec[:, t+1]
      syn[sidx] = al * so + h1v;
      h1out[sidx] = (__bf16)h1v;
    }
  }
}

// ------------- readout GEMM: h2(T,B,O) = spk_rec[:, :T] @ w2 ----------------
// grid = BT/64 = 1000, block = 128; N = 32 (O padded), 2 accumulators/wave.
// A operand needs f32->bf16 conversion in flight, so it keeps the VGPR path.

__global__ __launch_bounds__(128) void readout_gemm_kernel(
    const float*  __restrict__ spk_rec,  // [B][501][H]
    const __bf16* __restrict__ w2t,      // [OPAD][H]
    float*        __restrict__ h2)       // [T][B][O]
{
  __shared__ __align__(16) __bf16 Al[64 * LS];
  __shared__ __align__(16) __bf16 Bl[32 * LS];
  const int Rbase = blockIdx.x * 64;     // r = b*T + t
  const int tid = threadIdx.x;
  const int lane = tid & 31, wv = tid >> 5;
  const int hs = lane >> 4, l15 = lane & 15;
  const int srow = tid >> 1, shh = tid & 1;

  v8f acc[2] = {};

  for (int k0 = 0; k0 < H_; k0 += 32) {
    __syncthreads();
    { // stage A: spikes f32 -> bf16 (exact: values are 0/1)
      int r = Rbase + srow;
      int bb = r / T_;
      int tt = r - bb * T_;
      const v4f* g = (const v4f*)(spk_rec + (size_t)bb * (TP1 * H_) +
                                  (size_t)tt * H_ + k0 + shh * 16);
      __bf16* s = Al + srow * LS + shh * 16;
#pragma unroll
      for (int q = 0; q < 4; q++) {
        v4f f = g[q];
        s[q * 4 + 0] = (__bf16)f.x;
        s[q * 4 + 1] = (__bf16)f.y;
        s[q * 4 + 2] = (__bf16)f.z;
        s[q * 4 + 3] = (__bf16)f.w;
      }
    }
    if (tid < 64) { // stage Bt: 32 n x 32 k
      const v8bf* g = (const v8bf*)(w2t + (size_t)(tid >> 1) * H_ + k0 + (tid & 1) * 16);
      v8bf* s = (v8bf*)(Bl + (tid >> 1) * LS + (tid & 1) * 16);
      s[0] = g[0]; s[1] = g[1];
    }
    __syncthreads();

    Fr16 a;
    a.h[0] = *(const v8bf*)(Al + (wv * 16 + l15) * LS + hs * 8);
    a.h[1] = *(const v8bf*)(Al + (wv * 16 + l15) * LS + 16 + hs * 8);
#pragma unroll
    for (int j = 0; j < 2; j++) {
      Fr16 b;
      b.h[0] = *(const v8bf*)(Bl + (j * 16 + l15) * LS + hs * 8);
      b.h[1] = *(const v8bf*)(Bl + (j * 16 + l15) * LS + 16 + hs * 8);
      acc[j] = wmma_bf16(a.v, b.v, acc[j]);
    }
  }

#pragma unroll
  for (int j = 0; j < 2; j++) {
    int o = j * 16 + l15;
    if (o < O_) {
#pragma unroll
      for (int v = 0; v < 8; v++) {
        int r = Rbase + wv * 16 + v + 8 * hs;
        int bb = r / T_;
        int tt = r - bb * T_;
        h2[(size_t)tt * (B_ * O_) + (size_t)bb * O_ + o] = acc[j][v];
      }
    }
  }
}

// ----------------- output leaky-integrator scan (B*O chains) ----------------

__global__ void out_scan_kernel(const float* __restrict__ h2,
                                const float* __restrict__ alpha_o,
                                const float* __restrict__ beta_o,
                                float* __restrict__ out_rec) {
  int idx = blockIdx.x * blockDim.x + threadIdx.x;
  if (idx >= B_ * O_) return;
  int bb = idx / O_, o = idx - bb * O_;
  float al = alpha_o[o], be = beta_o[o];
  float flt = 0.0f, out = 0.0f;
  float* orow = out_rec + (size_t)bb * (TP1 * O_);
  for (int t = 0; t < T_; t++) {
    float x = h2[(size_t)t * (B_ * O_) + idx];
    float nf = al * flt + x;
    float no = be * out + flt;     // uses OLD flt, per reference
    orow[(t + 1) * O_ + o] = no;
    flt = nf;
    out = no;
  }
}

// ----------------------------------------------------------------------------

extern "C" void kernel_launch(void* const* d_in, const int* in_sizes, int n_in,
                              void* d_out, int out_size, void* d_ws, size_t ws_size,
                              hipStream_t stream) {
  (void)in_sizes; (void)n_in; (void)out_size; (void)ws_size;

  const float* inputs  = (const float*)d_in[0];
  const float* w1      = (const float*)d_in[1];
  const float* v1      = (const float*)d_in[2];
  const float* w2      = (const float*)d_in[3];
  const float* alpha_h = (const float*)d_in[4];
  const float* beta_h  = (const float*)d_in[5];
  const float* alpha_o = (const float*)d_in[6];
  const float* beta_o  = (const float*)d_in[7];

  float* out_rec = (float*)d_out;
  float* mem_rec = out_rec + (size_t)B_ * TP1 * O_;
  float* spk_rec = mem_rec + (size_t)B_ * TP1 * H_;

  char* ws = (char*)d_ws;
  size_t off = 0;
  auto alloc = [&](size_t bytes) -> void* {
    void* p = ws + off;
    off += (bytes + 255) & ~(size_t)255;
    return p;
  };
  __bf16* xbf = (__bf16*)alloc((size_t)B_ * T_ * IPAD * 2);   // ~90 MB (L2-resident)
  __bf16* w1t = (__bf16*)alloc((size_t)H_ * IPAD * 2);        // 1.4 MB
  __bf16* v1t = (__bf16*)alloc((size_t)H_ * H_ * 2);          // 2.1 MB (L2-resident)
  __bf16* w2t = (__bf16*)alloc((size_t)OPAD * H_ * 2);        // 64 KB
  __bf16* h1a = (__bf16*)alloc((size_t)B_ * H_ * 2);          // ping
  __bf16* h1b = (__bf16*)alloc((size_t)B_ * H_ * 2);          // pong
  float*  syn = (float*)alloc((size_t)B_ * H_ * 4);
  float*  h2  = (float*)alloc((size_t)T_ * B_ * O_ * 4);      // 5.1 MB
  float*  xft = (float*)alloc((size_t)T_ * B_ * H_ * 4);      // 262 MB

  init_kernel<<<(B_ * H_ + 255) / 256, 256, 0, stream>>>(
      out_rec, mem_rec, spk_rec, syn, h1a);

  long long nX = (long long)B_ * T_ * IPAD;
  cvt_inputs_kernel<<<(unsigned)((nX + 255) / 256), 256, 0, stream>>>(inputs, xbf);
  cvt_w1_kernel<<<(H_ * IPAD + 255) / 256, 256, 0, stream>>>(w1, w1t);
  cvt_v1_kernel<<<(H_ * H_ + 255) / 256, 256, 0, stream>>>(v1, v1t);
  cvt_w2_kernel<<<(OPAD * H_ + 255) / 256, 256, 0, stream>>>(w2, w2t);

  // Feedforward drive for all timesteps: 16000 workgroups.
  ff_gemm_kernel<<<(B_ * T_ / 64) * (H_ / 64), 128, 0, stream>>>(xbf, w1t, xft);

  // Sequential recurrence: one small fused GEMM+LIF kernel per step.
  for (int t = 0; t < T_; t++) {
    const __bf16* hin  = (t & 1) ? h1b : h1a;
    __bf16*       hout = (t & 1) ? h1a : h1b;
    rec_step_kernel<<<(B_ / 64) * (H_ / 64), 128, 0, stream>>>(
        t, xft, v1t, hin, hout, syn, alpha_h, beta_h, mem_rec, spk_rec);
  }

  readout_gemm_kernel<<<B_ * T_ / 64, 128, 0, stream>>>(spk_rec, w2t, h2);
  out_scan_kernel<<<(B_ * O_ + 255) / 256, 256, 0, stream>>>(
      h2, alpha_o, beta_o, out_rec);
}
